// Super1DLayer_18631568130115
// MI455X (gfx1250) — matrix operand order repaired and checked by reading, not compiled, not run on hardware
//
#include <hip/hip_runtime.h>

typedef __attribute__((ext_vector_type(16))) _Float16 v16h;
typedef __attribute__((ext_vector_type(8)))  _Float16 v8h;
typedef __attribute__((ext_vector_type(4)))  _Float16 v4h;
typedef __attribute__((ext_vector_type(8)))  float    v8f;

#define B_    4
#define C_    128
#define T_    2048
#define R_    1024
#define N_    32
#define WSTRIDE 56          // padded f16 row stride (112B: 16B-aligned, conflict-free)
#define EPS_  1e-10f

__global__ __launch_bounds__(256) void roi_gauss_pool_wmma(
    const float* __restrict__ x,      // [B, C, T]
    const float* __restrict__ rois,   // [R, 3]
    const float* __restrict__ center, // [N]
    const float* __restrict__ gamma,  // [N]
    float* __restrict__ out)          // [R, C, N]
{
    __shared__ _Float16 wtile[2][N_][WSTRIDE]; // double-buffered f16 weight tile [n][t-in-chunk]
    __shared__ float    ssum[N_][8];
    __shared__ float    sinv[N_];

    const int r     = blockIdx.x;
    const int tid   = threadIdx.x;
    const int lane  = tid & 31;
    const int wv    = tid >> 5;       // wave id 0..7 -> C-tile
    const int nw    = tid >> 3;       // n index for weight phase, 0..31
    const int slice = tid & 7;        // t-slice within chunk for weight phase

    // ROI parameters (uniform per block -> scalar loads)
    const int   bidx  = (int)rois[r * 3 + 0];
    const float start = rois[r * 3 + 1];
    const float end   = rois[r * 3 + 2];

    // Per-thread Gaussian params for its n
    const float mu = start + center[nw] * (end - start);
    const float gg = gamma[nw];
    const float g2 = gg * gg;

    const float* xb = x + (size_t)bidx * C_ * T_;

    // WMMA fragment addressing (wave32, 16x16x32 f16):
    //  A (16x32, M=c K=t): lane m = lane&15; lanes<16 hold K 0-7 & 16-23, lanes>=16 hold K 8-15 & 24-31
    //  B (32x16, K=t N=n): lane n = lane&15; lanes<16 hold K 0-7 & 16-23, lanes>=16 hold K 8-15 & 24-31
    const int mrow = lane & 15;
    const int kb0  = (lane < 16) ? 0 : 8;    // A fragment K base
    const int kbB  = (lane < 16) ? 0 : 8;    // B fragment K base
    const float* arow = xb + (size_t)(wv * 16 + mrow) * T_ + kb0;

    v8f acc0 = {};   // columns n = 0..15
    v8f acc1 = {};   // columns n = 16..31
    float sacc = 0.0f;

    for (int chunk = 0; chunk < T_ / 32; ++chunk) {
        const int tbase = chunk * 32;
        const int buf   = chunk & 1;

        // ---- Phase 1: compute 4 Gaussian weights (f32), accumulate sum, stash f16 in LDS ----
        {
            const float t0 = (float)(tbase + slice * 4);
            const float d0 = t0        - mu;
            const float d1 = t0 + 1.0f - mu;
            const float d2 = t0 + 2.0f - mu;
            const float d3 = t0 + 3.0f - mu;
            const float e0 = __expf(-g2 * d0 * d0);
            const float e1 = __expf(-g2 * d1 * d1);
            const float e2 = __expf(-g2 * d2 * d2);
            const float e3 = __expf(-g2 * d3 * d3);
            sacc += (e0 + e1) + (e2 + e3);
            v4h w4 = { (_Float16)e0, (_Float16)e1, (_Float16)e2, (_Float16)e3 };
            *(v4h*)&wtile[buf][nw][slice * 4] = w4;   // ds_store_b64
        }

        // ---- A fragment: global f32 -> f16 (4x global_load_b128, L2-resident) ----
        const float4 f0 = *(const float4*)(arow + tbase);        // K kb0+0..3
        const float4 f1 = *(const float4*)(arow + tbase + 4);    // K kb0+4..7
        const float4 f2 = *(const float4*)(arow + tbase + 16);   // K kb0+16..19
        const float4 f3 = *(const float4*)(arow + tbase + 20);   // K kb0+20..23
        v16h a;
        a[0]  = (_Float16)f0.x; a[1]  = (_Float16)f0.y;
        a[2]  = (_Float16)f0.z; a[3]  = (_Float16)f0.w;
        a[4]  = (_Float16)f1.x; a[5]  = (_Float16)f1.y;
        a[6]  = (_Float16)f1.z; a[7]  = (_Float16)f1.w;
        a[8]  = (_Float16)f2.x; a[9]  = (_Float16)f2.y;
        a[10] = (_Float16)f2.z; a[11] = (_Float16)f2.w;
        a[12] = (_Float16)f3.x; a[13] = (_Float16)f3.y;
        a[14] = (_Float16)f3.z; a[15] = (_Float16)f3.w;

        __syncthreads();   // weights for this chunk visible to all waves

        // ---- Phase 2: B fragments from LDS, two WMMAs ----
        const v8h blo0 = *(const v8h*)&wtile[buf][mrow][kbB];          // ds_load_b128
        const v8h bhi0 = *(const v8h*)&wtile[buf][mrow][kbB + 16];
        const v8h blo1 = *(const v8h*)&wtile[buf][16 + mrow][kbB];
        const v8h bhi1 = *(const v8h*)&wtile[buf][16 + mrow][kbB + 16];
        const v16h b0 = __builtin_shufflevector(blo0, bhi0,
            0,1,2,3,4,5,6,7,8,9,10,11,12,13,14,15);
        const v16h b1 = __builtin_shufflevector(blo1, bhi1,
            0,1,2,3,4,5,6,7,8,9,10,11,12,13,14,15);

        acc0 = __builtin_amdgcn_wmma_f32_16x16x32_f16(
            false, a, false, b0, (short)0, acc0, false, false);
        acc1 = __builtin_amdgcn_wmma_f32_16x16x32_f16(
            false, a, false, b1, (short)0, acc1, false, false);
        // double buffer: next chunk writes buf^1; hazard covered by next iter's barrier
    }

    // ---- Normalization scalars: S[n] = sum_t w[n,t];  inv = 1/(S + EPS) ----
    ssum[nw][slice] = sacc;
    __syncthreads();
    if (tid < N_) {
        float s = EPS_;
        #pragma unroll
        for (int j = 0; j < 8; ++j) s += ssum[tid][j];
        sinv[tid] = 1.0f / s;
    }
    __syncthreads();

    // ---- Epilogue: scale + store. C/D layout: VGPR v -> row v (lanes<16) / v+8 (lanes>=16) ----
    const float inv0 = sinv[mrow];
    const float inv1 = sinv[16 + mrow];
    const int rowoff = (lane < 16) ? 0 : 8;
    float* orow = out + (size_t)r * C_ * N_ + (size_t)(wv * 16 + rowoff) * N_ + mrow;
    #pragma unroll
    for (int v = 0; v < 8; ++v) {
        orow[(size_t)v * N_]      = acc0[v] * inv0;
        orow[(size_t)v * N_ + 16] = acc1[v] * inv1;
    }
}

extern "C" void kernel_launch(void* const* d_in, const int* in_sizes, int n_in,
                              void* d_out, int out_size, void* d_ws, size_t ws_size,
                              hipStream_t stream) {
    const float* x      = (const float*)d_in[0];
    const float* rois   = (const float*)d_in[1];
    const float* center = (const float*)d_in[2];
    const float* gamma  = (const float*)d_in[3];
    // d_in[4] = feature_dim scalar (N=32), compile-time constant here.
    float* out = (float*)d_out;

    const int R = in_sizes[1] / 3;   // 1024
    roi_gauss_pool_wmma<<<R, 256, 0, stream>>>(x, rois, center, gamma, out);
}
